// TrimKVQwen3ForCausalLM_3607772529318
// MI455X (gfx1250) — matrix operand order repaired and checked by reading, not compiled, not run on hardware
//
#include <hip/hip_runtime.h>
#include <hip/hip_bf16.h>

// ---------------------------------------------------------------------------
// Problem constants (from the reference)
// ---------------------------------------------------------------------------
constexpr int T   = 2048;
constexpr int HID = 2048;
constexpr int HQ  = 16;
constexpr int HKV = 8;
constexpr int D   = 128;
#define ATTN_SCALE 0.08838834764831845f   // 1/sqrt(128)

typedef __attribute__((ext_vector_type(16))) __bf16 v16bf;
typedef __attribute__((ext_vector_type(8)))  float  v8f;

// ---------------------------------------------------------------------------
// Helpers
// ---------------------------------------------------------------------------
__device__ __forceinline__ unsigned short f2bf(float f) {
    union { float f; unsigned int u; } c; c.f = f;
    unsigned int u = c.u;
    unsigned int r = u + 0x7FFFu + ((u >> 16) & 1u);   // round-to-nearest-even
    return (unsigned short)(r >> 16);
}

// A fragment (16x32, 16-bit): lanes 0-15 row M=lane hold K 0..15,
// lanes 16-31 hold K 16..31.  Source row-major [rows][ld] bf16.
__device__ __forceinline__ v16bf load_a16x32(const unsigned short* base, int ld,
                                             int row0, int k0, int lane) {
    int hf = lane >> 4, r = lane & 15;
    const unsigned short* p = base + (size_t)(row0 + r) * ld + k0 + hf * 16;
    union { uint4 u[2]; v16bf v; } x;
    x.u[0] = *(const uint4*)(p);        // K +0..7   (or +16..23)
    x.u[1] = *(const uint4*)(p + 8);    // K +8..15  (or +24..31)
    return x.v;
}

// B fragment (32x16, 16-bit): lanes 0-15 column N=lane hold K 0..15,
// lanes 16-31 hold K 16..31.  Source = B-transposed, row-major [N][ld] bf16.
__device__ __forceinline__ v16bf load_b32x16(const unsigned short* base, int ld,
                                             int n0, int k0, int lane) {
    int kh = lane >> 4, n = lane & 15;
    const unsigned short* p = base + (size_t)(n0 + n) * ld + k0 + kh * 16;
    union { uint4 u[2]; v16bf v; } x;
    x.u[0] = *(const uint4*)(p);        // K +0..7
    x.u[1] = *(const uint4*)(p + 8);    // K +8..15
    return x.v;
}

#define WMMA_BF16(a, b, c) \
    __builtin_amdgcn_wmma_f32_16x16x32_bf16(false, (a), false, (b), (short)0, (c), false, false)

// ---------------------------------------------------------------------------
// Elementwise / layout kernels
// ---------------------------------------------------------------------------
__global__ void k_f32_to_bf16(const float* __restrict__ in,
                              unsigned short* __restrict__ out, int n) {
    int i = blockIdx.x * 256 + threadIdx.x;
    if (i < n) out[i] = f2bf(in[i]);
}

// W (Kr x Nc, f32, row-major) -> Wt (Nc x Kr, bf16, row-major)
__global__ void k_cvt_transpose(const float* __restrict__ W,
                                unsigned short* __restrict__ Wt, int Kr, int Nc) {
    int i = blockIdx.x * 256 + threadIdx.x;
    if (i >= Kr * Nc) return;
    int k = i / Nc, n = i - k * Nc;
    Wt[(size_t)n * Kr + k] = f2bf(W[i]);
}

// V (T x HKV*D f32) -> Vt (HKV x D x T bf16)
__global__ void k_v_transpose(const float* __restrict__ V,
                              unsigned short* __restrict__ Vt) {
    int i = blockIdx.x * 256 + threadIdx.x;      // i < T*HKV*D
    int t = i >> 10;                             // / (HKV*D)
    int rem = i & 1023;
    int h = rem >> 7, d = rem & 127;
    Vt[(size_t)h * D * T + (size_t)d * T + t] = f2bf(V[i]);
}

// logb[h*T + t] = max(log(sigmoid(X[t]·gate_w[:,h] + gate_b[h])), log(1e-8))
__global__ __launch_bounds__(128) void k_gate(const float* __restrict__ X,
                                              const float* __restrict__ gw,
                                              const float* __restrict__ gb,
                                              float* __restrict__ logb) {
    int wid  = blockIdx.x * 4 + (threadIdx.x >> 5);
    int lane = threadIdx.x & 31;
    int h = wid & (HKV - 1), t = wid >> 3;
    float s = 0.f;
    for (int j = lane; j < HID; j += 32)
        s += X[(size_t)t * HID + j] * gw[(size_t)j * HKV + h];
    #pragma unroll
    for (int m = 16; m; m >>= 1) s += __shfl_xor(s, m, 32);
    if (lane == 0) {
        float x = s + gb[h];
        float ls = (x >= 0.f) ? -log1pf(__expf(-x)) : (x - log1pf(__expf(x)));
        logb[(size_t)h * T + t] = fmaxf(ls, -18.420681f);
    }
}

// Per-head RMSNorm + RoPE: in (T x H*D f32) -> out (H x T x D bf16)
__global__ __launch_bounds__(128) void k_norm_rope(const float* __restrict__ in,
                                                   unsigned short* __restrict__ out,
                                                   const float* __restrict__ nw,
                                                   const float* __restrict__ cs,
                                                   const float* __restrict__ sn,
                                                   int H) {
    int wid  = blockIdx.x * 4 + (threadIdx.x >> 5);
    int lane = threadIdx.x & 31;
    int t = wid / H, h = wid - t * H;
    const float* p = in + (size_t)t * H * D + (size_t)h * D;
    float x[4];
    #pragma unroll
    for (int i = 0; i < 4; ++i) x[i] = p[lane + 32 * i];
    float ss = x[0]*x[0] + x[1]*x[1] + x[2]*x[2] + x[3]*x[3];
    #pragma unroll
    for (int m = 16; m; m >>= 1) ss += __shfl_xor(ss, m, 32);
    float inv = rsqrtf(ss * (1.0f / D) + 1e-6f);
    float xn[4];
    #pragma unroll
    for (int i = 0; i < 4; ++i) {
        int d = lane + 32 * i;
        xn[i] = x[i] * inv * nw[d];
    }
    unsigned short* q = out + (size_t)h * T * D + (size_t)t * D;
    #pragma unroll
    for (int i = 0; i < 4; ++i) {
        int d = lane + 32 * i;
        float rot = (i < 2) ? -xn[i + 2] : xn[i - 2];   // rotate_half partner: d +/- 64
        float v = xn[i] * cs[(size_t)t * D + d] + rot * sn[(size_t)t * D + d];
        q[d] = f2bf(v);
    }
}

// ---------------------------------------------------------------------------
// Generic bf16 WMMA GEMM:  C(f32, MxN) = A(bf16, MxK) * Bt(bf16, NxK)^T
// One wave computes a 32x64 C tile (2 A frags x 4 B frags -> 8 wmma/k-step).
// All 12 b128 loads for a k-step are issued as one clause before the wmma
// group so s_wait_loadcnt staggers instead of stalling each wmma.
// ---------------------------------------------------------------------------
__global__ __launch_bounds__(128) void k_gemm_bf16(const unsigned short* __restrict__ A,
                                                   const unsigned short* __restrict__ Bt,
                                                   float* __restrict__ C,
                                                   int M, int N, int K) {
    int lane = threadIdx.x & 31, w = threadIdx.x >> 5;
    int m0 = blockIdx.y * 128 + w * 32;
    int n0 = blockIdx.x * 64;
    v8f acc[2][4] = {};
    for (int k0 = 0; k0 < K; k0 += 32) {
        if (k0 + 32 < K) {
            __builtin_prefetch(A  + (size_t)(m0 + (lane & 15)) * K + k0 + 32, 0, 3);
            __builtin_prefetch(Bt + (size_t)(n0 + (lane & 15)) * K + k0 + 32, 0, 3);
        }
        // ---- issue every fragment load up front ----
        v16bf a0 = load_a16x32(A, K, m0,      k0, lane);
        v16bf a1 = load_a16x32(A, K, m0 + 16, k0, lane);
        v16bf b[4];
        #pragma unroll
        for (int j = 0; j < 4; ++j)
            b[j] = load_b32x16(Bt, K, n0 + j * 16, k0, lane);
        // ---- then the wmma group ----
        #pragma unroll
        for (int j = 0; j < 4; ++j)
            acc[0][j] = WMMA_BF16(a0, b[j], acc[0][j]);
        #pragma unroll
        for (int j = 0; j < 4; ++j)
            acc[1][j] = WMMA_BF16(a1, b[j], acc[1][j]);
    }
    int hf = lane >> 4, nn = lane & 15;
    #pragma unroll
    for (int sm = 0; sm < 2; ++sm)
        #pragma unroll
        for (int j = 0; j < 4; ++j)
            #pragma unroll
            for (int r = 0; r < 8; ++r)
                C[(size_t)(m0 + sm * 16 + r + 8 * hf) * N + (n0 + j * 16 + nn)] = acc[sm][j][r];
}

// ---------------------------------------------------------------------------
// Flash-style gated causal attention.
// One wave per (q-head, 16-query block).  Keys processed in chunks of 32.
//   Qbf: HQ  x T x D bf16     Kbf: HKV x T x D bf16
//   Vt : HKV x D x T bf16     logb: HKV x T f32
//   Obf: T x (HQ*D) bf16   (attention output, pre out-projection)
// ---------------------------------------------------------------------------
__global__ __launch_bounds__(128) void k_attn(const unsigned short* __restrict__ Qbf,
                                              const unsigned short* __restrict__ Kbf,
                                              const unsigned short* __restrict__ Vt,
                                              const float* __restrict__ logb,
                                              unsigned short* __restrict__ Obf) {
    __shared__ float          sc[4][16][32];   // staged scores (f32)
    __shared__ unsigned short pb[4][16][32];   // staged probabilities (bf16)
    __shared__ float          fac[4][16];      // per-row rescale factors
    int wq = threadIdx.x >> 5, lane = threadIdx.x & 31;
    int wid = blockIdx.x * 4 + wq;
    int h  = wid >> 7;            // q head
    int qb = wid & 127;           // query block
    int hk = h >> 1;              // kv head (G=2)
    int qbase = qb * 16;
    const unsigned short* Qh = Qbf + (size_t)h  * T * D;
    const unsigned short* Kh = Kbf + (size_t)hk * T * D;
    const unsigned short* Vh = Vt  + (size_t)hk * D * T;
    const float* lb = logb + (size_t)hk * T;
    int hf = lane >> 4, nn = lane & 15;

    // Resident Q fragments covering D=128 (4 x K32)
    v16bf aq[4];
    #pragma unroll
    for (int c = 0; c < 4; ++c) aq[c] = load_a16x32(Qh, D, qbase, c * 32, lane);

    v8f o[8] = {};                          // 16 x 128 f32 output accumulator
    float runmax = -1e30f, runsum = 0.f;    // valid on lanes 0-15 (row = lane)
    int kend = qbase + 16;                  // causal: keys j <= i < kend

    for (int k0 = 0; k0 < kend; k0 += 32) {
        // ---- scores: S = Q * K^T.  Load all 8 K frags, then 8 wmma ----
        v16bf bk[8];
        #pragma unroll
        for (int c = 0; c < 4; ++c) {
            bk[c]     = load_b32x16(Kh, D, k0,      c * 32, lane);
            bk[4 + c] = load_b32x16(Kh, D, k0 + 16, c * 32, lane);
        }
        v8f s0 = {}, s1 = {};
        #pragma unroll
        for (int c = 0; c < 4; ++c) s0 = WMMA_BF16(aq[c], bk[c],     s0);
        #pragma unroll
        for (int c = 0; c < 4; ++c) s1 = WMMA_BF16(aq[c], bk[4 + c], s1);

        // stage to LDS in row-major [m][key] layout
        #pragma unroll
        for (int r = 0; r < 8; ++r) {
            sc[wq][r + 8 * hf][nn]      = s0[r];
            sc[wq][r + 8 * hf][16 + nn] = s1[r];
        }
        __asm__ volatile("s_wait_dscnt 0x0" ::: "memory");

        // ---- streaming softmax with retention bias: one row per lane ----
        if (lane < 16) {
            int i = qbase + lane;
            float vals[32];
            float mx = -1e30f;
            #pragma unroll
            for (int jj = 0; jj < 32; ++jj) {
                int key = k0 + jj;
                float v = -1e30f;
                if (key <= i)
                    v = sc[wq][lane][jj] * ATTN_SCALE + lb[key] * (float)(i - key);
                vals[jj] = v;
                mx = fmaxf(mx, v);
            }
            float nm = fmaxf(runmax, mx);
            float f  = __expf(runmax - nm);
            float sum = 0.f;
            #pragma unroll
            for (int jj = 0; jj < 32; ++jj) {
                float pv = __expf(vals[jj] - nm);
                vals[jj] = pv;
                sum += pv;
            }
            runsum = runsum * f + sum;
            runmax = nm;
            fac[wq][lane] = f;
            unsigned int* prow = (unsigned int*)&pb[wq][lane][0];
            #pragma unroll
            for (int jj = 0; jj < 16; ++jj)
                prow[jj] = (unsigned int)f2bf(vals[2 * jj]) |
                           ((unsigned int)f2bf(vals[2 * jj + 1]) << 16);
        }
        __asm__ volatile("s_wait_dscnt 0x0" ::: "memory");

        // ---- rescale output accumulator, then O += P * V ----
        float fr[8];
        #pragma unroll
        for (int r = 0; r < 8; ++r) fr[r] = fac[wq][r + 8 * hf];
        #pragma unroll
        for (int j = 0; j < 8; ++j)
            #pragma unroll
            for (int r = 0; r < 8; ++r) o[j][r] *= fr[r];

        v16bf ap = load_a16x32(&pb[wq][0][0], 32, 0, 0, lane);   // P as A frag (ds_load_b128)
        v16bf bv[8];                                             // load all V frags first
        #pragma unroll
        for (int j = 0; j < 8; ++j)
            bv[j] = load_b32x16(Vh, T, j * 16, k0, lane);        // V^T (DxT) as Bt
        #pragma unroll
        for (int j = 0; j < 8; ++j)
            o[j] = WMMA_BF16(ap, bv[j], o[j]);
    }

    // ---- finalize: divide by running denom, store bf16 ----
    if (lane < 16) fac[wq][lane] = 1.0f / runsum;
    __asm__ volatile("s_wait_dscnt 0x0" ::: "memory");
    float fr[8];
    #pragma unroll
    for (int r = 0; r < 8; ++r) fr[r] = fac[wq][r + 8 * hf];
    #pragma unroll
    for (int j = 0; j < 8; ++j)
        #pragma unroll
        for (int r = 0; r < 8; ++r) {
            int row = qbase + r + 8 * hf;
            int col = h * D + j * 16 + nn;
            Obf[(size_t)row * (HQ * D) + col] = f2bf(o[j][r] * fr[r]);
        }
}

// ---------------------------------------------------------------------------
// Host launcher
// ---------------------------------------------------------------------------
extern "C" void kernel_launch(void* const* d_in, const int* in_sizes, int n_in,
                              void* d_out, int out_size, void* d_ws, size_t ws_size,
                              hipStream_t stream) {
    (void)in_sizes; (void)n_in; (void)out_size; (void)ws_size;
    const float* X    = (const float*)d_in[0];
    const float* Wq   = (const float*)d_in[1];
    const float* Wk   = (const float*)d_in[2];
    const float* Wv   = (const float*)d_in[3];
    const float* Wo   = (const float*)d_in[4];
    const float* qnw  = (const float*)d_in[5];
    const float* knw  = (const float*)d_in[6];
    const float* gw   = (const float*)d_in[7];
    const float* gb   = (const float*)d_in[8];
    const float* cosp = (const float*)d_in[9];
    const float* sinp = (const float*)d_in[10];
    float* out = (float*)d_out;
    char*  ws  = (char*)d_ws;

    // workspace layout (bytes, all 256-aligned)
    constexpr size_t o_xbf  = 0;                          // T*HID bf16       8 MiB
    constexpr size_t o_wqt  = o_xbf  + (size_t)T * HID * 2;
    constexpr size_t o_wkt  = o_wqt  + (size_t)HID * HQ  * D * 2;
    constexpr size_t o_wvt  = o_wkt  + (size_t)HID * HKV * D * 2;
    constexpr size_t o_wot  = o_wvt  + (size_t)HID * HKV * D * 2;
    constexpr size_t o_qf   = o_wot  + (size_t)HID * HQ  * D * 2;
    constexpr size_t o_kf   = o_qf   + (size_t)T * HQ  * D * 4;
    constexpr size_t o_vf   = o_kf   + (size_t)T * HKV * D * 4;
    constexpr size_t o_qbf  = o_vf   + (size_t)T * HKV * D * 4;
    constexpr size_t o_kbf  = o_qbf  + (size_t)T * HQ  * D * 2;
    constexpr size_t o_vt   = o_kbf  + (size_t)T * HKV * D * 2;
    constexpr size_t o_obf  = o_vt   + (size_t)T * HKV * D * 2;
    constexpr size_t o_logb = o_obf  + (size_t)T * HQ  * D * 2;

    unsigned short* Xbf  = (unsigned short*)(ws + o_xbf);
    unsigned short* Wqt  = (unsigned short*)(ws + o_wqt);
    unsigned short* Wkt  = (unsigned short*)(ws + o_wkt);
    unsigned short* Wvt  = (unsigned short*)(ws + o_wvt);
    unsigned short* Wot  = (unsigned short*)(ws + o_wot);
    float*          Qf   = (float*)(ws + o_qf);
    float*          Kf   = (float*)(ws + o_kf);
    float*          Vf   = (float*)(ws + o_vf);
    unsigned short* Qbfp = (unsigned short*)(ws + o_qbf);
    unsigned short* Kbfp = (unsigned short*)(ws + o_kbf);
    unsigned short* Vtp  = (unsigned short*)(ws + o_vt);
    unsigned short* Obfp = (unsigned short*)(ws + o_obf);
    float*          logb = (float*)(ws + o_logb);

    // 1) activations + weights -> bf16 (weights transposed to [N][K])
    k_f32_to_bf16 <<<(T * HID) / 256, 256, 0, stream>>>(X, Xbf, T * HID);
    k_cvt_transpose<<<(HID * 2048) / 256, 256, 0, stream>>>(Wq, Wqt, HID, 2048);
    k_cvt_transpose<<<(HID * 1024) / 256, 256, 0, stream>>>(Wk, Wkt, HID, 1024);
    k_cvt_transpose<<<(HID * 1024) / 256, 256, 0, stream>>>(Wv, Wvt, HID, 1024);
    k_cvt_transpose<<<(2048 * HID) / 256, 256, 0, stream>>>(Wo, Wot, 2048, HID);

    // 2) retention gate -> log(beta)
    k_gate<<<(T * HKV) / 4, 128, 0, stream>>>(X, gw, gb, logb);

    // 3) QKV projections (bf16 WMMA, f32 accumulate)
    {
        dim3 g(2048 / 64, 2048 / 128);
        k_gemm_bf16<<<g, 128, 0, stream>>>(Xbf, Wqt, Qf, 2048, 2048, 2048);
    }
    {
        dim3 g(1024 / 64, 2048 / 128);
        k_gemm_bf16<<<g, 128, 0, stream>>>(Xbf, Wkt, Kf, 2048, 1024, 2048);
        k_gemm_bf16<<<g, 128, 0, stream>>>(Xbf, Wvt, Vf, 2048, 1024, 2048);
    }

    // 4) per-head RMSNorm + RoPE (fp32), emit head-major bf16; V -> [h][d][t]
    k_norm_rope<<<(T * HQ)  / 4, 128, 0, stream>>>(Qf, Qbfp, qnw, cosp, sinp, HQ);
    k_norm_rope<<<(T * HKV) / 4, 128, 0, stream>>>(Kf, Kbfp, knw, cosp, sinp, HKV);
    k_v_transpose<<<(T * HKV * D) / 256, 256, 0, stream>>>(Vf, Vtp);

    // 5) gated causal flash attention
    k_attn<<<(HQ * (T / 16)) / 4, 128, 0, stream>>>(Qbfp, Kbfp, Vtp, logb, Obfp);

    // 6) output projection -> d_out (f32)
    {
        dim3 g(2048 / 64, 2048 / 128);
        k_gemm_bf16<<<g, 128, 0, stream>>>(Obfp, Wot, out, 2048, 2048, 2048);
    }
}
